// GCN_17016660427224
// MI455X (gfx1250) — matrix operand order repaired and checked by reading, not compiled, not run on hardware
//
#include <hip/hip_runtime.h>
#include <hip/hip_bf16.h>

typedef __attribute__((ext_vector_type(16))) _Float16 v16h;
typedef __attribute__((ext_vector_type(8)))  float    v8f;

#define N_NODES   20000
#define IN_FEAT   128
#define CONV_OUT  32
#define N_EDGES   640000
#define FC1_OUT   128
#define FLAT_SZ   (N_NODES * CONV_OUT)   // 640000

// ---------------- init kernels ----------------

__global__ void gcn_init_deg(float* deg) {
    int i = blockIdx.x * blockDim.x + threadIdx.x;
    if (i < N_NODES) deg[i] = 1.0f;   // self-loop contributes 1
}

__global__ void gcn_deg_count(const int* __restrict__ dst, float* deg) {
    int e = blockIdx.x * blockDim.x + threadIdx.x;
    if (e < N_EDGES) atomicAdd(&deg[dst[e]], 1.0f);
}

__global__ void gcn_dinv(float* deg) {
    int i = blockIdx.x * blockDim.x + threadIdx.x;
    if (i < N_NODES) deg[i] = rsqrtf(deg[i]);   // deg >= 1 always
}

__global__ void gcn_init_agg(const float* __restrict__ b_conv, float* agg) {
    int i = blockIdx.x * blockDim.x + threadIdx.x;
    if (i < FLAT_SZ) agg[i] = b_conv[i & (CONV_OUT - 1)];
}

__global__ void gcn_init_h3(float* h3raw) {
    if (threadIdx.x < FC1_OUT) h3raw[threadIdx.x] = 0.0f;
}

// ---------------- WMMA GEMM: h = x @ w_conv  ([20000,128] x [128,32]) ----------------
// 1 wave computes one 16x16 tile of h. grid = (1250 M-tiles, 2 N-tiles).
// K=128 -> 4 x v_wmma_f32_16x16x32_f16 (f32 accumulate).

__global__ __launch_bounds__(32) void gcn_gemm_wmma(const float* __restrict__ x,
                                                    const float* __restrict__ w,
                                                    float* __restrict__ h) {
    const int mtile = blockIdx.x;          // 0..1249
    const int ntile = blockIdx.y;          // 0..1
    const int lane  = threadIdx.x;         // 0..31 (wave32)
    const int m     = lane & 15;           // row (A) / col (B) role of lane
    const int khalf = lane >> 4;           // lane-half selects K sub-range

    const int arow = mtile * 16 + m;       // x row this lane feeds
    const int bcol = ntile * 16 + m;       // w col this lane feeds

    v8f acc = {};
    #pragma unroll
    for (int k0 = 0; k0 < IN_FEAT; k0 += 32) {
        v16h a, b;
        // A fragment: 16-bit A 16x32 layout (ISA 7.12.2):
        // VGPR v: K = (v>>2)*16 + khalf*8 + (v&3)*2  (+0,+1 packed)
        #pragma unroll
        for (int v = 0; v < 8; ++v) {
            int kof = k0 + ((v >> 2) * 16) + khalf * 8 + ((v & 3) * 2);
            a[2 * v]     = (_Float16)x[arow * IN_FEAT + kof];
            a[2 * v + 1] = (_Float16)x[arow * IN_FEAT + kof + 1];
        }
        // B fragment: same K order as A: e<8 -> K=khalf*8+e, e>=8 -> K=16+khalf*8+(e-8)
        #pragma unroll
        for (int e = 0; e < 16; ++e) {
            int kk = k0 + ((e >> 3) * 16) + khalf * 8 + (e & 7);
            b[e] = (_Float16)w[kk * CONV_OUT + bcol];
        }
        acc = __builtin_amdgcn_wmma_f32_16x16x32_f16(
            /*neg_a=*/false, a, /*neg_b=*/false, b,
            /*c_mod=*/(short)0, acc, /*reuse_a=*/false, /*reuse_b=*/false);
    }
    // C/D 32-bit layout: VGPR r -> M = r + khalf*8, N = lane&15
    #pragma unroll
    for (int r = 0; r < 8; ++r) {
        int row = mtile * 16 + r + khalf * 8;
        int col = ntile * 16 + m;
        h[row * CONV_OUT + col] = acc[r];
    }
}

// ---------------- edge scatter: agg[dst] += h[src] * dinv[src]*dinv[dst] ----------------
// lane = channel -> fully coalesced gather + coalesced L2 atomics (h/agg fit in L2).

__global__ __launch_bounds__(256) void gcn_scatter(const int* __restrict__ src,
                                                   const int* __restrict__ dst,
                                                   const float* __restrict__ dinv,
                                                   const float* __restrict__ h,
                                                   float* agg) {
    int idx = blockIdx.x * blockDim.x + threadIdx.x;
    int e = idx >> 5;
    int c = idx & 31;
    const int total = N_EDGES + N_NODES;
    if (e >= total) return;
    int s, d;
    if (e < N_EDGES) { s = src[e]; d = dst[e]; }
    else             { s = d = e - N_EDGES; }       // self-loop
    float norm = dinv[s] * dinv[d];
    atomicAdd(&agg[d * CONV_OUT + c], h[s * CONV_OUT + c] * norm);
}

// ---------------- fc1: h3raw[j] = sum_i relu(agg[i]) * w_fc1[j,i] ----------------
// 327 MB of w_fc1 streamed exactly once; agg (2.56 MB) re-read from L2.

#define FC1_CHUNKS 160
#define FC1_PER    (FLAT_SZ / FC1_CHUNKS)   // 4000

__global__ __launch_bounds__(256) void gcn_fc1(const float* __restrict__ agg,
                                               const float* __restrict__ w1,
                                               float* h3raw) {
    const int j     = blockIdx.y;                 // output feature 0..127
    const int i0    = blockIdx.x * FC1_PER;
    const float* wrow = w1 + (size_t)j * FLAT_SZ;
    float sum = 0.0f;
    for (int i = i0 + threadIdx.x; i < i0 + FC1_PER; i += 256) {
        float a = agg[i];
        a = a > 0.0f ? a : 0.0f;                  // fused ReLU of h1
        sum += a * wrow[i];
    }
    __shared__ float sm[256];
    sm[threadIdx.x] = sum;
    __syncthreads();
    #pragma unroll
    for (int s = 128; s > 0; s >>= 1) {
        if (threadIdx.x < s) sm[threadIdx.x] += sm[threadIdx.x + s];
        __syncthreads();
    }
    if (threadIdx.x == 0) atomicAdd(&h3raw[j], sm[0]);
}

// ---------------- fc2: out[i] = relu( relu(h3raw+b1) . w_fc2[i,:] + b2[i] ) ----------------
// one wave32 per output row, lanes cover 128 weights (4 each), shfl_xor reduce.

__global__ __launch_bounds__(256) void gcn_fc2(const float* __restrict__ h3raw,
                                               const float* __restrict__ b1,
                                               const float* __restrict__ w2,
                                               const float* __restrict__ b2,
                                               float* __restrict__ out) {
    __shared__ float h3[FC1_OUT];
    if (threadIdx.x < FC1_OUT) {
        float v = h3raw[threadIdx.x] + b1[threadIdx.x];
        h3[threadIdx.x] = v > 0.0f ? v : 0.0f;
    }
    __syncthreads();
    const int lane = threadIdx.x & 31;
    const int wv   = threadIdx.x >> 5;
    const int i    = blockIdx.x * 8 + wv;
    if (i >= N_NODES) return;
    float sum = 0.0f;
    #pragma unroll
    for (int k = 0; k < 4; ++k)
        sum += h3[lane + 32 * k] * w2[(size_t)i * FC1_OUT + lane + 32 * k];
    #pragma unroll
    for (int off = 16; off > 0; off >>= 1)
        sum += __shfl_xor(sum, off, 32);
    if (lane == 0) {
        float v = sum + b2[i];
        out[i] = v > 0.0f ? v : 0.0f;
    }
}

// ---------------- launch ----------------

extern "C" void kernel_launch(void* const* d_in, const int* in_sizes, int n_in,
                              void* d_out, int out_size, void* d_ws, size_t ws_size,
                              hipStream_t stream) {
    const float* x      = (const float*)d_in[0];
    const int*   eidx   = (const int*)d_in[1];          // [2, 640000]: row0=src, row1=dst
    const float* w_conv = (const float*)d_in[2];
    const float* b_conv = (const float*)d_in[3];
    const float* w_fc1  = (const float*)d_in[4];
    const float* b_fc1  = (const float*)d_in[5];
    const float* w_fc2  = (const float*)d_in[6];
    const float* b_fc2  = (const float*)d_in[7];
    float* out = (float*)d_out;

    const int* e_src = eidx;
    const int* e_dst = eidx + N_EDGES;

    // workspace layout (floats)
    float* ws    = (float*)d_ws;
    float* deg   = ws;                        // 20000 (becomes dinv)
    float* h     = deg + N_NODES;             // 640000
    float* agg   = h + FLAT_SZ;               // 640000
    float* h3raw = agg + FLAT_SZ;             // 128

    gcn_init_deg <<<(N_NODES + 255) / 256, 256, 0, stream>>>(deg);
    gcn_deg_count<<<(N_EDGES + 255) / 256, 256, 0, stream>>>(e_dst, deg);
    gcn_dinv     <<<(N_NODES + 255) / 256, 256, 0, stream>>>(deg);
    gcn_init_agg <<<(FLAT_SZ + 255) / 256, 256, 0, stream>>>(b_conv, agg);

    gcn_gemm_wmma<<<dim3(N_NODES / 16, CONV_OUT / 16), 32, 0, stream>>>(x, w_conv, h);

    {
        long long threads = (long long)(N_EDGES + N_NODES) * 32;
        int blocks = (int)((threads + 255) / 256);
        gcn_scatter<<<blocks, 256, 0, stream>>>(e_src, e_dst, deg, h, agg);
    }

    gcn_init_h3<<<1, FC1_OUT, 0, stream>>>(h3raw);
    gcn_fc1<<<dim3(FC1_CHUNKS, FC1_OUT), 256, 0, stream>>>(agg, w_fc1, h3raw);
    gcn_fc2<<<(N_NODES + 7) / 8, 256, 0, stream>>>(h3raw, b_fc1, w_fc2, b_fc2, out);
}
